// MoELayer_1159641170595
// MI455X (gfx1250) — matrix operand order repaired and checked by reading, not compiled, not run on hardware
//
#include <hip/hip_runtime.h>

// ---------------------------------------------------------------------------
// Problem constants (match reference)
// ---------------------------------------------------------------------------
static constexpr int Bc   = 4;
static constexpr int Sc   = 512;
static constexpr int Hc   = 1024;
static constexpr int NHc  = 16;
static constexpr int HDc  = 64;
static constexpr int DFFc = 4096;
static constexpr int Ec   = 4;
static constexpr int Tc   = Bc * Sc;          // 2048 tokens

// ---------------------------------------------------------------------------
// Types
// ---------------------------------------------------------------------------
typedef __bf16 v16bf __attribute__((ext_vector_type(16)));
typedef __bf16 v8bf  __attribute__((ext_vector_type(8)));
typedef float  v8f   __attribute__((ext_vector_type(8)));
typedef unsigned int u32x4 __attribute__((ext_vector_type(4)));
typedef int          i32x8 __attribute__((ext_vector_type(8)));
typedef int          i32x4 __attribute__((ext_vector_type(4)));

// Native bf16 convert (v_cvt_*bf16_f32 on gfx1250) instead of manual rounding.
__device__ __forceinline__ unsigned short f2bf(float f) {
  return __builtin_bit_cast(unsigned short, (__bf16)f);
}

__device__ __forceinline__ v8f wmma_bf16(v16bf a, v16bf b, v8f c) {
  return __builtin_amdgcn_wmma_f32_16x16x32_bf16(false, a, false, b, (short)0, c,
                                                 false, false);
}

__device__ __forceinline__ v16bf concat8(v8bf lo, v8bf hi) {
  return __builtin_shufflevector(lo, hi, 0, 1, 2, 3, 4, 5, 6, 7,
                                 8, 9, 10, 11, 12, 13, 14, 15);
}

// ---------------------------------------------------------------------------
// WMMA fragment loaders (CDNA5 layouts, ISA 7.12.2) — vectorized to b128.
// A 16x32 bf16: lane L holds row M=L%16; K in {8g..8g+7} U {16+8g..16+8g+7},
// g = L/16. Two contiguous 8-half runs -> two 128-bit loads.
// ---------------------------------------------------------------------------
__device__ __forceinline__ v16bf load_a_rowmajor(const unsigned short* base,
                                                 int ld, int lane) {
  int m = lane & 15;
  int g = (lane >> 4) * 8;
  const __bf16* r = (const __bf16*)(base + (size_t)m * ld + g);
  v8bf lo = *(const v8bf*)(r);
  v8bf hi = *(const v8bf*)(r + 16);
  return concat8(lo, hi);
}

// B 32x16 bf16 with the contraction dim contiguous in memory:
// element (k, n) at base[n*ld + k]; lane n = L%16 needs K in {8g..8g+7} U
// {16+8g..16+8g+7}, g = L/16 -> two 128-bit loads.
__device__ __forceinline__ v16bf load_b_colmajor(const unsigned short* base,
                                                 int ld, int lane) {
  int n  = lane & 15;
  int kb = (lane >> 4) * 8;
  const __bf16* c = (const __bf16*)(base + (size_t)n * ld + kb);
  v8bf lo = *(const v8bf*)(c);
  v8bf hi = *(const v8bf*)(c + 16);
  return concat8(lo, hi);
}

// ---------------------------------------------------------------------------
// Tensor Data Mover: 2D bf16 tile (tile_h x tile_w) -> LDS, row stride
// `stride_elems` (elements). Descriptor per D# group0/group1 bit layout.
// 6-arg builtin: (u32x4 g0, i32x8 g1, i32x4 g2, i32x4 g3, i32x8 pad, i32 cpol)
// ---------------------------------------------------------------------------
#if __has_builtin(__builtin_amdgcn_tensor_load_to_lds) && \
    __has_builtin(__builtin_amdgcn_s_wait_tensorcnt)
#define USE_TDM 1
#else
#define USE_TDM 0
#endif

#if USE_TDM
__device__ __forceinline__ void tdm_load_tile_bf16(unsigned int lds_off,
                                                   const void* gptr,
                                                   unsigned int row_elems,
                                                   unsigned int rows,
                                                   unsigned int tile_w,
                                                   unsigned int tile_h,
                                                   unsigned int stride_elems) {
  unsigned long long ga = (unsigned long long)(uintptr_t)gptr;
  u32x4 g0;
  g0[0] = 1u;                                   // count=1, user descriptor
  g0[1] = lds_off;                              // lds_addr
  g0[2] = (unsigned int)(ga & 0xFFFFFFFFull);
  g0[3] = (unsigned int)((ga >> 32) & 0x01FFFFFFull) | (2u << 30); // type=2
  i32x8 g1;
  g1[0] = (int)(1u << 16);                      // data_size=1 (2 bytes)
  g1[1] = (int)((row_elems & 0xFFFFu) << 16);   // tensor_dim0[15:0]
  g1[2] = (int)(((row_elems >> 16) & 0xFFFFu) | ((rows & 0xFFFFu) << 16));
  g1[3] = (int)(((rows >> 16) & 0xFFFFu) | ((tile_w & 0xFFFFu) << 16));
  g1[4] = (int)(tile_h & 0xFFFFu);              // tile_dim1 (tile_dim2 = 0)
  g1[5] = (int)stride_elems;                    // tensor_dim0_stride[31:0]
  g1[6] = 0;
  g1[7] = 0;
  i32x4 z4 = {0, 0, 0, 0};
  i32x8 z8 = {0, 0, 0, 0, 0, 0, 0, 0};
  __builtin_amdgcn_tensor_load_to_lds(g0, g1, z4, z4, z8, 0);
}
#endif

// ---------------------------------------------------------------------------
// GEMM: C[M,N](f32) = A[M,K](bf16 bits) @ W[K,N](f32, converted to bf16)
// mode 0: C = acc
// mode 1: C = acc + res
// mode 3: C += gate[row] * (acc + res)        (top-k combine epilogue)
// Block: 256 threads / 8 waves; tile 64x128, K-step 32.
// Waves: wm = wave/2 (16-row chunk), wn = wave%2 (64-col chunk);
// each wave computes 16x64 = 4 accumulators -> 4 WMMA per K-step.
// A tile staged row-major via TDM; W tile staged COLUMN-major (lB[n*BK+k]).
// ---------------------------------------------------------------------------
__global__ void gemm_bf16_kernel(const unsigned short* __restrict__ A,
                                 const float* __restrict__ W,
                                 float* __restrict__ C,
                                 const float* __restrict__ res,
                                 const float* __restrict__ gate,
                                 int M, int N, int K, int mode) {
  constexpr int BM = 64, BN = 128, BK = 32;
  __shared__ unsigned short lA[BM * BK];        // row-major [m][k]
  __shared__ unsigned short lB[BN * BK];        // column-major [n][k]

  const int tid  = threadIdx.x;
  const int lane = tid & 31;
  const int wave = tid >> 5;
  const int wm   = wave >> 1;                   // 0..3 (16-row chunk)
  const int wn   = wave & 1;                    // 0..1 (64-col chunk)
  const int bm   = blockIdx.x * BM;
  const int bn   = blockIdx.y * BN;

  v8f zf = {0, 0, 0, 0, 0, 0, 0, 0};
  v8f acc[4] = {zf, zf, zf, zf};

  for (int k0 = 0; k0 < K; k0 += BK) {
    __syncthreads();                            // protect LDS reuse (WAR)

#if USE_TDM
    if (wave == 0) {
      tdm_load_tile_bf16((unsigned int)(uintptr_t)lA,
                         A + (size_t)bm * K + k0,
                         (unsigned int)K, (unsigned int)M,
                         BK, BM, (unsigned int)K);
    }
#else
    {
      int i = tid * 8;                          // 2048 halves / 256 thr
      int r = i >> 5, c = i & 31;
      *(uint4*)&lA[i] = *(const uint4*)&A[(size_t)(bm + r) * K + k0 + c];
    }
#endif
    // W tile fp32 -> bf16, transposed into LDS (coalesced float4 reads)
#pragma unroll
    for (int it = 0; it < 4; ++it) {
      int i = (tid + it * 256) * 4;             // element index in 32x128 tile
      int kk = i >> 7;                          // row (k)
      int nn = i & 127;                         // col (multiple of 4)
      float4 w4 = *(const float4*)&W[(size_t)(k0 + kk) * N + bn + nn];
      lB[(nn + 0) * BK + kk] = f2bf(w4.x);
      lB[(nn + 1) * BK + kk] = f2bf(w4.y);
      lB[(nn + 2) * BK + kk] = f2bf(w4.z);
      lB[(nn + 3) * BK + kk] = f2bf(w4.w);
    }
    if (k0 + BK < K)                            // hint next weight tile
      __builtin_prefetch(&W[(size_t)(k0 + BK + (tid >> 5)) * N + bn + 4 * (tid & 31)], 0, 1);
#if USE_TDM
    if (wave == 0) __builtin_amdgcn_s_wait_tensorcnt(0);
#endif
    __syncthreads();

    v16bf af = load_a_rowmajor(&lA[(wm * 16) * BK], BK, lane);
#pragma unroll
    for (int t = 0; t < 4; ++t) {
      v16bf bf = load_b_colmajor(&lB[(wn * 64 + 16 * t) * BK], BK, lane);
      acc[t] = wmma_bf16(af, bf, acc[t]);
    }
  }

  const int g16 = lane >> 4;
  const int n   = lane & 15;
  const int row0 = bm + wm * 16 + 8 * g16;
  const int col0 = bn + wn * 64 + n;
#pragma unroll
  for (int r = 0; r < 8; ++r) {
    int row = row0 + r;
    float gt = (mode == 3) ? gate[row] : 0.0f;
#pragma unroll
    for (int t = 0; t < 4; ++t) {
      size_t i0 = (size_t)row * N + col0 + 16 * t;
      float v0 = acc[t][r];
      if (mode == 1) {
        C[i0] = v0 + res[i0];
      } else if (mode == 3) {
        C[i0] += gt * (v0 + res[i0]);
      } else {
        C[i0] = v0;
      }
    }
  }
}

// ---------------------------------------------------------------------------
// Flash-style causal attention, one (b,head) per blockIdx.y,
// 128 query rows per block (16 per wave). HD=64. Online softmax.
// K is read via col-major fragments (hd contiguous); V is pre-transposed to
// [B][H][HD][S] so the key (contraction) dim is contiguous too.
// ---------------------------------------------------------------------------
__global__ void attn_kernel(const unsigned short* __restrict__ qb,
                            const unsigned short* __restrict__ kbuf,
                            const unsigned short* __restrict__ vT,
                            unsigned short* __restrict__ ob) {
  __shared__ unsigned short lP[8 * 16 * 32];    // per-wave P tile (16x32)
  const int tid  = threadIdx.x;
  const int lane = tid & 31;
  const int wave = tid >> 5;
  const int bh = blockIdx.y;
  const int b  = bh >> 4;                       // NH = 16
  const int hh = bh & 15;
  const int q0 = blockIdx.x * 128 + wave * 16;

  const unsigned short* qbase = qb + ((size_t)(b * Sc + q0)) * Hc + hh * HDc;
  v16bf qa0 = load_a_rowmajor(qbase, Hc, lane);       // hd 0..31
  v16bf qa1 = load_a_rowmajor(qbase + 32, Hc, lane);  // hd 32..63

  float rmax[8], rsum[8];
  v8f zf = {0, 0, 0, 0, 0, 0, 0, 0};
  v8f oacc[4] = {zf, zf, zf, zf};
#pragma unroll
  for (int r = 0; r < 8; ++r) { rmax[r] = -1e30f; rsum[r] = 0.0f; }

  const int g16 = lane >> 4;
  const int n   = lane & 15;
  unsigned short* pbase = &lP[wave * 512];
  const unsigned short* vtbase = vT + (size_t)(b * Hc + hh * HDc) * Sc;

  for (int k0 = 0; k0 < q0 + 16; k0 += 32) {
    const unsigned short* kbb = kbuf + ((size_t)(b * Sc + k0)) * Hc + hh * HDc;

    v8f s0 = zf, s1 = zf;                       // keys k0..+15 / k0+16..+31
    s0 = wmma_bf16(qa0, load_b_colmajor(kbb, Hc, lane), s0);
    s0 = wmma_bf16(qa1, load_b_colmajor(kbb + 32, Hc, lane), s0);
    s1 = wmma_bf16(qa0, load_b_colmajor(kbb + (size_t)16 * Hc, Hc, lane), s1);
    s1 = wmma_bf16(qa1, load_b_colmajor(kbb + (size_t)16 * Hc + 32, Hc, lane), s1);

#pragma unroll
    for (int r = 0; r < 8; ++r) {
      int qrow = q0 + 8 * g16 + r;
      float v0 = s0[r] * 0.125f;                // 1/sqrt(64)
      float v1 = s1[r] * 0.125f;
      if (k0 + n > qrow)      v0 = -1e9f;
      if (k0 + 16 + n > qrow) v1 = -1e9f;
      float mx = fmaxf(v0, v1);
#pragma unroll
      for (int o = 1; o < 16; o <<= 1) mx = fmaxf(mx, __shfl_xor(mx, o, 32));
      float mnew = fmaxf(rmax[r], mx);
      float corr = __expf(rmax[r] - mnew);
      float p0 = __expf(v0 - mnew);
      float p1 = __expf(v1 - mnew);
      float ps = p0 + p1;
#pragma unroll
      for (int o = 1; o < 16; o <<= 1) ps += __shfl_xor(ps, o, 32);
      rsum[r] = rsum[r] * corr + ps;
      rmax[r] = mnew;
#pragma unroll
      for (int t = 0; t < 4; ++t) oacc[t][r] *= corr;
      int prow = 8 * g16 + r;
      pbase[prow * 32 + n]      = f2bf(p0);
      pbase[prow * 32 + n + 16] = f2bf(p1);
    }
    // reload P in A-fragment layout; P @ V (32-deep contraction, keys contig)
    v16bf pa = load_a_rowmajor(pbase, 32, lane);
#pragma unroll
    for (int t = 0; t < 4; ++t)
      oacc[t] = wmma_bf16(pa, load_b_colmajor(vtbase + (size_t)(16 * t) * Sc + k0,
                                              Sc, lane),
                          oacc[t]);
  }

  unsigned short* obase = ob + ((size_t)(b * Sc + q0)) * Hc + hh * HDc;
#pragma unroll
  for (int r = 0; r < 8; ++r) {
    float inv = 1.0f / rsum[r];
    int row = 8 * g16 + r;
#pragma unroll
    for (int t = 0; t < 4; ++t)
      obase[(size_t)row * Hc + 16 * t + n] = f2bf(oacc[t][r] * inv);
  }
}

// ---------------------------------------------------------------------------
// RMSNorm -> bf16 bits. One block per token.
// ---------------------------------------------------------------------------
__global__ void rmsnorm_kernel(const float* __restrict__ x,
                               const float* __restrict__ w,
                               unsigned short* __restrict__ out) {
  __shared__ float red[256];
  __shared__ float sscale;
  const int t = blockIdx.x, tid = threadIdx.x;
  float s = 0.0f;
  for (int i = tid; i < Hc; i += 256) {
    float v = x[(size_t)t * Hc + i];
    s += v * v;
  }
  red[tid] = s;
  __syncthreads();
  for (int o = 128; o > 0; o >>= 1) {
    if (tid < o) red[tid] += red[tid + o];
    __syncthreads();
  }
  if (tid == 0) sscale = rsqrtf(red[0] * (1.0f / Hc) + 1e-6f);
  __syncthreads();
  float sc = sscale;
  for (int i = tid; i < Hc; i += 256) {
    size_t idx = (size_t)t * Hc + i;
    out[idx] = f2bf(x[idx] * sc * w[i]);
  }
}

// ---------------------------------------------------------------------------
// RoPE for q/k (+ bf16 pack); v packed bf16 AND transposed to [B][H][HD][S].
// ---------------------------------------------------------------------------
__global__ void rope_cvt_kernel(const float* __restrict__ q,
                                const float* __restrict__ k,
                                const float* __restrict__ v,
                                const float* __restrict__ cosT,
                                const float* __restrict__ sinT,
                                unsigned short* __restrict__ qb,
                                unsigned short* __restrict__ kb,
                                unsigned short* __restrict__ vT) {
  int idx = blockIdx.x * 256 + threadIdx.x;     // < Tc*Hc
  int t = idx >> 10;                            // / Hc
  int c = idx & (Hc - 1);
  int bidx = t >> 9;                            // / Sc
  int spos = t & (Sc - 1);
  int d = c & (HDc - 1);
  float cv = cosT[spos * HDc + d];
  float sv = sinT[spos * HDc + d];
  int pidx = (d < HDc / 2) ? idx + HDc / 2 : idx - HDc / 2;
  float sgn = (d < HDc / 2) ? -1.0f : 1.0f;
  qb[idx] = f2bf(q[idx] * cv + sgn * q[pidx] * sv);
  kb[idx] = f2bf(k[idx] * cv + sgn * k[pidx] * sv);
  vT[((size_t)bidx * Hc + c) * Sc + spos] = f2bf(v[idx]);
}

// ---------------------------------------------------------------------------
// SwiGLU fuse: mb = bf16( silu(g) * u )
// ---------------------------------------------------------------------------
__global__ void silu_mul_kernel(const float* __restrict__ g,
                                const float* __restrict__ u,
                                unsigned short* __restrict__ mb) {
  size_t idx = (size_t)blockIdx.x * 256 + threadIdx.x;  // < Tc*DFFc
  float gv = g[idx];
  float s = gv * (1.0f / (1.0f + __expf(-gv)));
  mb[idx] = f2bf(s * u[idx]);
}

// ---------------------------------------------------------------------------
// Router: logits, softmax, top-2 gates; deterministic per-block partial sums
// of probs (for P) and selection counts (for f).
// ---------------------------------------------------------------------------
__global__ void router_kernel(const float* __restrict__ h,
                              const float* __restrict__ rw,
                              float* __restrict__ gates,      // [E][T]
                              float* __restrict__ partial) {  // [nblk][8]
  __shared__ float red[256];
  const int tid = threadIdx.x;
  const int t = blockIdx.x * 256 + tid;
  float lg[Ec] = {0, 0, 0, 0};
  for (int i = 0; i < Hc; ++i) {
    float hv = h[(size_t)t * Hc + i];
#pragma unroll
    for (int e = 0; e < Ec; ++e) lg[e] += hv * rw[i * Ec + e];
  }
  float m = fmaxf(fmaxf(lg[0], lg[1]), fmaxf(lg[2], lg[3]));
  float p[Ec], s = 0.0f;
#pragma unroll
  for (int e = 0; e < Ec; ++e) { p[e] = __expf(lg[e] - m); s += p[e]; }
  float inv = 1.0f / s;
#pragma unroll
  for (int e = 0; e < Ec; ++e) p[e] *= inv;
  int i0 = 0;
#pragma unroll
  for (int e = 1; e < Ec; ++e) if (p[e] > p[i0]) i0 = e;
  int i1 = (i0 == 0) ? 1 : 0;
#pragma unroll
  for (int e = 0; e < Ec; ++e) if (e != i0 && p[e] > p[i1]) i1 = e;
  float wsum = p[i0] + p[i1];
#pragma unroll
  for (int e = 0; e < Ec; ++e) {
    float gw = (e == i0) ? p[i0] / wsum : (e == i1) ? p[i1] / wsum : 0.0f;
    gates[(size_t)e * Tc + t] = gw;
  }
  float vals[8];
#pragma unroll
  for (int e = 0; e < Ec; ++e) {
    vals[e] = p[e];
    vals[4 + e] = (e == i0 || e == i1) ? 1.0f : 0.0f;
  }
  for (int q = 0; q < 8; ++q) {
    red[tid] = vals[q];
    __syncthreads();
    for (int o = 128; o > 0; o >>= 1) {
      if (tid < o) red[tid] += red[tid + o];
      __syncthreads();
    }
    if (tid == 0) partial[blockIdx.x * 8 + q] = red[0];
    __syncthreads();
  }
}

__global__ void loss_kernel(const float* __restrict__ partial,
                            float* __restrict__ loss_out, int nblk) {
  if (threadIdx.x == 0 && blockIdx.x == 0) {
    float P[Ec] = {0, 0, 0, 0}, F[Ec] = {0, 0, 0, 0};
    for (int b = 0; b < nblk; ++b)
      for (int e = 0; e < Ec; ++e) {
        P[e] += partial[b * 8 + e];
        F[e] += partial[b * 8 + 4 + e];
      }
    float loss = 0.0f;
    for (int e = 0; e < Ec; ++e)
      loss += (F[e] / (float)Tc) * (P[e] / (float)Tc);
    loss_out[0] = (float)Ec * loss;
  }
}

__global__ void zero_kernel(float* __restrict__ out) {
  size_t idx = (size_t)blockIdx.x * 256 + threadIdx.x;
  out[idx] = 0.0f;
}

// ---------------------------------------------------------------------------
// Launch
// ---------------------------------------------------------------------------
extern "C" void kernel_launch(void* const* d_in, const int* in_sizes, int n_in,
                              void* d_out, int out_size, void* d_ws,
                              size_t ws_size, hipStream_t stream) {
  (void)in_sizes; (void)n_in; (void)out_size; (void)ws_size;
  const float* h    = (const float*)d_in[0];
  const float* cosT = (const float*)d_in[1];
  const float* sinT = (const float*)d_in[2];
  const float* rw   = (const float*)d_in[3];
  const float* wq   = (const float*)d_in[4];
  const float* wk   = (const float*)d_in[5];
  const float* wv   = (const float*)d_in[6];
  const float* wo   = (const float*)d_in[7];
  const float* wg   = (const float*)d_in[8];
  const float* wu   = (const float*)d_in[9];
  const float* wd   = (const float*)d_in[10];
  const float* n1   = (const float*)d_in[11];
  const float* n2   = (const float*)d_in[12];
  float* out = (float*)d_out;

  // workspace carve-up
  char* ws = (char*)d_ws;
  size_t off = 0;
  auto carve = [&](size_t bytes) -> void* {
    void* p = ws + off;
    off = (off + bytes + 255) & ~(size_t)255;
    return p;
  };
  float*          gates   = (float*)carve((size_t)Ec * Tc * 4);
  float*          partial = (float*)carve((size_t)(Tc / 256) * 8 * 4);
  unsigned short* xb = (unsigned short*)carve((size_t)Tc * Hc * 2);
  float*          qf = (float*)carve((size_t)Tc * Hc * 4);
  float*          kf = (float*)carve((size_t)Tc * Hc * 4);
  float*          vf = (float*)carve((size_t)Tc * Hc * 4);
  unsigned short* qb2 = (unsigned short*)carve((size_t)Tc * Hc * 2);
  unsigned short* kb2 = (unsigned short*)carve((size_t)Tc * Hc * 2);
  unsigned short* vT  = (unsigned short*)carve((size_t)Tc * Hc * 2);
  unsigned short* ob = (unsigned short*)carve((size_t)Tc * Hc * 2);
  float*          h1 = (float*)carve((size_t)Tc * Hc * 4);
  unsigned short* yb = (unsigned short*)carve((size_t)Tc * Hc * 2);
  float*          gf = (float*)carve((size_t)Tc * DFFc * 4);
  float*          uf = (float*)carve((size_t)Tc * DFFc * 4);
  unsigned short* mb = (unsigned short*)carve((size_t)Tc * DFFc * 2);

  dim3 blk(256);

  zero_kernel<<<(Tc * Hc) / 256, blk, 0, stream>>>(out);
  router_kernel<<<Tc / 256, blk, 0, stream>>>(h, rw, gates, partial);

  for (int e = 0; e < Ec; ++e) {
    const float* wq_e = wq + (size_t)e * Hc * Hc;
    const float* wk_e = wk + (size_t)e * Hc * Hc;
    const float* wv_e = wv + (size_t)e * Hc * Hc;
    const float* wo_e = wo + (size_t)e * Hc * Hc;
    const float* wg_e = wg + (size_t)e * Hc * DFFc;
    const float* wu_e = wu + (size_t)e * Hc * DFFc;
    const float* wd_e = wd + (size_t)e * DFFc * Hc;
    const float* n1_e = n1 + (size_t)e * Hc;
    const float* n2_e = n2 + (size_t)e * Hc;

    rmsnorm_kernel<<<Tc, blk, 0, stream>>>(h, n1_e, xb);

    dim3 gQKV(Tc / 64, Hc / 128);
    gemm_bf16_kernel<<<gQKV, blk, 0, stream>>>(xb, wq_e, qf, nullptr, nullptr,
                                               Tc, Hc, Hc, 0);
    gemm_bf16_kernel<<<gQKV, blk, 0, stream>>>(xb, wk_e, kf, nullptr, nullptr,
                                               Tc, Hc, Hc, 0);
    gemm_bf16_kernel<<<gQKV, blk, 0, stream>>>(xb, wv_e, vf, nullptr, nullptr,
                                               Tc, Hc, Hc, 0);

    rope_cvt_kernel<<<(Tc * Hc) / 256, blk, 0, stream>>>(qf, kf, vf, cosT, sinT,
                                                         qb2, kb2, vT);

    dim3 gAttn(Sc / 128, Bc * NHc);
    attn_kernel<<<gAttn, blk, 0, stream>>>(qb2, kb2, vT, ob);

    gemm_bf16_kernel<<<gQKV, blk, 0, stream>>>(ob, wo_e, h1, h, nullptr,
                                               Tc, Hc, Hc, 1);

    rmsnorm_kernel<<<Tc, blk, 0, stream>>>(h1, n2_e, yb);

    dim3 gFF(Tc / 64, DFFc / 128);
    gemm_bf16_kernel<<<gFF, blk, 0, stream>>>(yb, wg_e, gf, nullptr, nullptr,
                                              Tc, DFFc, Hc, 0);
    gemm_bf16_kernel<<<gFF, blk, 0, stream>>>(yb, wu_e, uf, nullptr, nullptr,
                                              Tc, DFFc, Hc, 0);

    silu_mul_kernel<<<(Tc * DFFc) / 256, blk, 0, stream>>>(gf, uf, mb);

    dim3 gDown(Tc / 64, Hc / 128);
    gemm_bf16_kernel<<<gDown, blk, 0, stream>>>(mb, wd_e, out, h1,
                                                gates + (size_t)e * Tc,
                                                Tc, Hc, DFFc, 3);
  }

  loss_kernel<<<1, 32, 0, stream>>>(partial, out + (size_t)Tc * Hc, Tc / 256);
}
